// QuantizedLinear_31241592111614
// MI455X (gfx1250) — matrix operand order repaired and checked by reading, compiled, not run on hardware
//
#include <hip/hip_runtime.h>

// ---------------------------------------------------------------------------
// QuantizedLinear (AQLM-style) for MI455X / gfx1250.
//   out[8192, 4096] = x[8192, 4096] @ W.T[4096, 4096] + bias
//   W[o, g*8+j] = codebooks[codes[o,g], j] * scales[o]
//
// Stage 1: dequantize W -> bf16 in d_ws (4096*4096*2 = 32 MiB scratch).
// Stage 2: tiled GEMM with v_wmma_f32_16x16x32_bf16 (wave32, fp32 accum).
//          B (weight) tiles streamed memory->LDS with
//          GLOBAL_LOAD_ASYNC_TO_LDS_B128 (ASYNCcnt), A (x) tiles converted
//          fp32->bf16 through VGPRs.
// ---------------------------------------------------------------------------

typedef __attribute__((ext_vector_type(16))) __bf16 v16bf;
typedef __attribute__((ext_vector_type(8)))  __bf16 v8bf;
typedef __attribute__((ext_vector_type(8)))  float  v8f;
typedef __attribute__((ext_vector_type(4)))  float  v4f;

#define IN_F   4096
#define OUT_F  4096
#define M_ROWS 8192     // 4 * 2048
#define NIG    512      // IN_F / 8
#define IGS    8

#define BM 128
#define BN 128
#define BK 32
#define LDSS 40         // LDS row stride in bf16 elems (80 B, keeps 16B align)
#define NK (IN_F / BK)  // 128 K-steps

// ---------------------------------------------------------------------------
// Stage 1: codebook gather + scale -> bf16 weight matrix in workspace.
// ---------------------------------------------------------------------------
__global__ __launch_bounds__(256) void dequant_kernel(
    const int* __restrict__ codes,      // [4096, 512, 1]
    const float* __restrict__ cb,       // [1, 65536, 1, 8]
    const float* __restrict__ scales,   // [4096]
    void* __restrict__ Wv)              // out: bf16 [4096, 4096]
{
    __bf16* W = (__bf16*)Wv;
    const int idx = blockIdx.x * 256 + threadIdx.x;   // 0 .. 4096*512-1
    const int o = idx >> 9;
    const int g = idx & 511;

    const int code = codes[idx];
    const v4f* p = (const v4f*)(cb + ((size_t)code << 3));
    const v4f c0 = p[0];
    const v4f c1 = p[1];
    const float s = scales[o];

    v8bf w;
#pragma unroll
    for (int j = 0; j < 4; ++j) w[j]     = (__bf16)(c0[j] * s);
#pragma unroll
    for (int j = 0; j < 4; ++j) w[4 + j] = (__bf16)(c1[j] * s);

    *(v8bf*)(W + (size_t)o * IN_F + g * IGS) = w;
}

// ---------------------------------------------------------------------------
// Stage 2: C = X * W^T + bias with bf16 WMMA.
// Workgroup: 256 threads = 8 waves, computes a 128x128 C tile.
// Wave grid 4(M) x 2(N): each wave owns 32 rows x 64 cols = 2x4 WMMA tiles.
// Double-buffered LDS.
// ---------------------------------------------------------------------------

// Single merged LDS block so the raw byte offsets handed to the async-to-LDS
// instructions are well defined (single __shared__ object -> LDS offset 0).
//   A buf b : bf16 index  b*BM*LDSS
//   B buf b : bf16 index (2*BM + b*BN)*LDSS
#define A_IDX(b)  ((b) * BM * LDSS)
#define B_IDX(b)  ((2 * BM + (b) * BN) * LDSS)

__global__ __launch_bounds__(256) void wmma_gemm_kernel(
    const float* __restrict__ X,        // [8192, 4096] fp32
    const void*  __restrict__ Wv,       // [4096, 4096] bf16 (row = out feature)
    const float* __restrict__ bias,     // [4096]
    float* __restrict__ out)            // [8192, 4096] fp32
{
    const __bf16* W = (const __bf16*)Wv;

    __shared__ __bf16 smem[(2 * BM + 2 * BN) * LDSS]; // 40960 B

    const int tid   = threadIdx.x;
    const int lane  = tid & 31;
    const int wid   = tid >> 5;
    const int waveM = wid & 3;          // 0..3 -> 32-row slice
    const int waveN = wid >> 2;         // 0..1 -> 64-col slice
    const int lm    = lane & 15;
    const int lh    = lane >> 4;        // lane half (ISA 16-bit A/B layout)

    const int rowBase = blockIdx.y * BM;
    const int colBase = blockIdx.x * BN;

    // global->LDS staging role: each thread owns one tile row half (16 elems)
    const int ldr = tid >> 1;           // 0..127
    const int ldh = tid & 1;            // 0/1 -> K sub-chunk of 16

    const float*  xg = X + (size_t)(rowBase + ldr) * IN_F + ldh * 16;
    const __bf16* wg = W + (size_t)(colBase + ldr) * IN_F + ldh * 16;

    v4f ax[4];                           // 16 fp32 from x

    auto loadA = [&](int k0) {
        const v4f* xp = (const v4f*)(xg + k0);
        ax[0] = xp[0]; ax[1] = xp[1]; ax[2] = xp[2]; ax[3] = xp[3];
    };
    auto storeA = [&](int buf) {
        __bf16* ad = &smem[A_IDX(buf) + ldr * LDSS + ldh * 16];
        v8bf lo, hi;
#pragma unroll
        for (int j = 0; j < 8; ++j) lo[j] = (__bf16)ax[j >> 2][j & 3];
#pragma unroll
        for (int j = 0; j < 8; ++j) hi[j] = (__bf16)ax[2 + (j >> 2)][j & 3];
        ((v8bf*)ad)[0] = lo;
        ((v8bf*)ad)[1] = hi;
    };
    // Async DMA of this thread's 32B of the weight tile straight into LDS.
    // Tracked by ASYNCcnt; LDS VGPR operand is the byte address within the
    // wave's LDS allocation.
    auto asyncB = [&](int buf, int k0) {
        const __bf16* gp = wg + k0;
        unsigned l0 = (unsigned)((B_IDX(buf) + ldr * LDSS + ldh * 16) * 2);
        asm volatile("global_load_async_to_lds_b128 %0, %1, off"
                     :: "v"(l0), "v"(gp) : "memory");
        const __bf16* gp1 = gp + 8;
        unsigned l1 = l0 + 16;
        asm volatile("global_load_async_to_lds_b128 %0, %1, off"
                     :: "v"(l1), "v"(gp1) : "memory");
    };

    v8f acc[2][4];
#pragma unroll
    for (int mt = 0; mt < 2; ++mt)
#pragma unroll
        for (int nt = 0; nt < 4; ++nt)
            acc[mt][nt] = v8f{};

    // prologue: stage K-step 0
    asyncB(0, 0);
    loadA(0);
    storeA(0);
    asm volatile("s_wait_asynccnt 0" ::: "memory");
    __syncthreads();

    // A-matrix 16-bit layout (ISA 7.12.2): lanes 0-15 hold K {0..7,16..23},
    // lanes 16-31 hold K {8..15,24..31}; M = lane%16.
    const int kA0 = lh ? 8 : 0;

    for (int kt = 0; kt < NK; ++kt) {
        const int buf = kt & 1;

        if (kt + 1 < NK) {
            asyncB(buf ^ 1, (kt + 1) * BK);   // DMA next W tile -> LDS
            loadA((kt + 1) * BK);             // next x tile -> VGPRs
        }
        if (kt + 2 < NK) {
            __builtin_prefetch(xg + (kt + 2) * BK, 0, 3);
            __builtin_prefetch(wg + (kt + 2) * BK, 0, 3);
        }

        v16bf af[2];
#pragma unroll
        for (int mt = 0; mt < 2; ++mt) {
            const __bf16* ap =
                &smem[A_IDX(buf) + (waveM * 32 + mt * 16 + lm) * LDSS + kA0];
            v8bf a0 = *(const v8bf*)ap;          // K = kA0 .. kA0+7
            v8bf a1 = *(const v8bf*)(ap + 16);   // K = kA0+16 .. kA0+23
            af[mt] = __builtin_shufflevector(
                a0, a1, 0, 1, 2, 3, 4, 5, 6, 7, 8, 9, 10, 11, 12, 13, 14, 15);
        }

        // B-matrix 16-bit layout: N = lane%16; lanes 0-15 hold K {0..7,16..23},
        // lanes 16-31 hold K {8..15,24..31} (same K order as A).
        v16bf bfv[4];
#pragma unroll
        for (int nt = 0; nt < 4; ++nt) {
            const __bf16* bp =
                &smem[B_IDX(buf) + (waveN * 64 + nt * 16 + lm) * LDSS + lh * 8];
            v8bf b0 = *(const v8bf*)bp;
            v8bf b1 = *(const v8bf*)(bp + 16);
            bfv[nt] = __builtin_shufflevector(
                b0, b1, 0, 1, 2, 3, 4, 5, 6, 7, 8, 9, 10, 11, 12, 13, 14, 15);
        }

#pragma unroll
        for (int mt = 0; mt < 2; ++mt)
#pragma unroll
            for (int nt = 0; nt < 4; ++nt)
                acc[mt][nt] = __builtin_amdgcn_wmma_f32_16x16x32_bf16(
                    false, af[mt], false, bfv[nt],
                    (short)0, acc[mt][nt], false, false);

        if (kt + 1 < NK) storeA(buf ^ 1);
        // All of this wave's async B-tile writes must land before the
        // workgroup barrier publishes the buffer.
        asm volatile("s_wait_asynccnt 0" ::: "memory");
        __syncthreads();
    }

    // Epilogue: C/D layout — VGPR i holds M = i (lanes 0-15) / 8+i (16-31),
    // N = lane%16.
#pragma unroll
    for (int nt = 0; nt < 4; ++nt) {
        const int col = colBase + waveN * 64 + nt * 16 + lm;
        const float bv = bias[col];
#pragma unroll
        for (int mt = 0; mt < 2; ++mt) {
            const int row0 = rowBase + waveM * 32 + mt * 16 + lh * 8;
#pragma unroll
            for (int i = 0; i < 8; ++i)
                out[(size_t)(row0 + i) * OUT_F + col] = acc[mt][nt][i] + bv;
        }
    }
}

// ---------------------------------------------------------------------------
extern "C" void kernel_launch(void* const* d_in, const int* in_sizes, int n_in,
                              void* d_out, int out_size, void* d_ws, size_t ws_size,
                              hipStream_t stream) {
    const float* x         = (const float*)d_in[0];  // [4,2048,4096]
    const int*   codes     = (const int*)d_in[1];    // [4096,512,1]
    const float* codebooks = (const float*)d_in[2];  // [1,65536,1,8]
    const float* scales    = (const float*)d_in[3];  // [4096,1,1,1]
    const float* bias      = (const float*)d_in[4];  // [4096]
    float*       out       = (float*)d_out;          // [4,2048,4096]

    // Workspace: bf16 dequantized weights, 4096*4096*2 = 32 MiB.
    void* Wbf16 = d_ws;

    // Stage 1: dequantize (2M threads, one per (out, group)).
    dequant_kernel<<<(OUT_F * NIG) / 256, 256, 0, stream>>>(
        codes, codebooks, scales, Wbf16);

    // Stage 2: WMMA GEMM. Grid exactly tiles 8192x4096 -> EXEC all-ones.
    dim3 grid(OUT_F / BN, M_ROWS / BM);  // (32, 64)
    wmma_gemm_kernel<<<grid, 256, 0, stream>>>(x, Wbf16, bias, out);
}
